// ACTPC_47957604827293
// MI455X (gfx1250) — matrix operand order repaired and checked by reading, not compiled, not run on hardware
//
#include <hip/hip_runtime.h>
#include <hip/hip_bf16.h>

#define BATCH 512
#define TSTEPS 128
#define DIN0 128
#define HID 256
#define NCLUST 64
#define OUTD 32
#define BT (BATCH * TSTEPS)

// Padded LDS leading dimensions (bank-conflict-free)
#define LDH 264    // bf16 rows (132 DWORDs, == 4 mod 64 banks)
#define LDG 1028   // f32 gate rows
#define LDF 260    // f32 MLP staging rows
#define LDL_SEL 68 // f32 selector logits
#define LDL_PRE 36 // f32 predictor logits

typedef __attribute__((ext_vector_type(16))) __bf16 v16bf;
typedef __attribute__((ext_vector_type(8)))  float  v8f;
typedef __attribute__((ext_vector_type(4)))  unsigned int u32x4;

// ---------------------------------------------------------------------------
// WMMA fragment helpers (layouts per CDNA5 ISA §7.12.2)
// ---------------------------------------------------------------------------

__device__ __forceinline__ v16bf frag_from2(const __bf16* p0, const __bf16* p1) {
  union { v16bf v; u32x4 q[2]; } u;
  u.q[0] = *(const u32x4*)p0;
  u.q[1] = *(const u32x4*)p1;
  return u.v;
}

__device__ __forceinline__ v16bf frag_from2_nt(const __bf16* p0, const __bf16* p1) {
  union { v16bf v; u32x4 q[2]; } u;
  u.q[0] = __builtin_nontemporal_load((const u32x4*)p0);
  u.q[1] = __builtin_nontemporal_load((const u32x4*)p1);
  return u.v;
}

// A-matrix fragment (16x32 bf16). Row-major source: 16 rows at base + r*stride.
// lane<16 -> row=lane, elems 0..7 = K 0..7, elems 8..15 = K 16..23
// lane>=16 -> row=lane-16, K offsets +8.
__device__ __forceinline__ v16bf load_a(const __bf16* base, size_t stride, int kt, int lane) {
  int r  = lane & 15;
  int ko = (lane < 16) ? 0 : 8;
  const __bf16* p = base + (size_t)r * stride + kt * 32 + ko;
  return frag_from2(p, p + 16);
}

// Non-temporal variant for streamed-once global A operands.
__device__ __forceinline__ v16bf load_a_nt(const __bf16* base, size_t stride, int kt, int lane) {
  int r  = lane & 15;
  int ko = (lane < 16) ? 0 : 8;
  const __bf16* p = base + (size_t)r * stride + kt * 32 + ko;
  return frag_from2_nt(p, p + 16);
}

// B-matrix fragment (32x16 bf16) from pre-packed weights: lane L's 16 elements
// contiguous at ((kt*NT+nt)*32 + L)*16.  Weights are reused -> default RT hint.
__device__ __forceinline__ v16bf load_b(const __bf16* Wp, int kt, int nt, int NT, int lane) {
  const __bf16* p = Wp + ((((size_t)kt * NT + nt) * 32 + lane) << 4);
  return frag_from2(p, p + 8);
}

__device__ __forceinline__ v8f wmma_bf16(v16bf a, v16bf b, v8f c) {
  return __builtin_amdgcn_wmma_f32_16x16x32_bf16(false, a, false, b, (short)0, c, false, false);
}

__device__ __forceinline__ v8f vzero() {
  v8f r;
#pragma unroll
  for (int i = 0; i < 8; ++i) r[i] = 0.f;
  return r;
}

// C/D 16x16 f32 layout: lane<16 -> col=lane, vgpr r -> row r; lane>=16 -> row 8+r.
__device__ __forceinline__ void store_c(float* dst, int ld, v8f c, int lane) {
  int col = lane & 15;
  int r0  = (lane < 16) ? 0 : 8;
#pragma unroll
  for (int r = 0; r < 8; ++r) dst[(size_t)(r0 + r) * ld + col] = c[r];
}

__device__ __forceinline__ float sigm(float x) { return 1.f / (1.f + __expf(-x)); }

// ---------------------------------------------------------------------------
// Prep: pack f32 weight (K x N row-major) into fragment-native bf16 layout
// ---------------------------------------------------------------------------
__global__ void pack_w_kernel(const float* __restrict__ W, __bf16* __restrict__ out,
                              int Kdim, int Ndim) {
  int p = blockIdx.x * blockDim.x + threadIdx.x;
  if (p >= Kdim * Ndim) return;
  int e    = p & 15;
  int L    = (p >> 4) & 31;
  int tile = p >> 9;
  int NT   = Ndim >> 4;
  int nt   = tile % NT;
  int kt   = tile / NT;
  int k = kt * 32 + ((L < 16) ? 0 : 8) + ((e < 8) ? e : e + 8);
  int n = nt * 16 + (L & 15);
  out[p] = (__bf16)W[(size_t)k * Ndim + n];
}

// ---------------------------------------------------------------------------
// Prep: mask (any nonzero feature) + x f32 -> bf16
// ---------------------------------------------------------------------------
__global__ void mask_cvt_kernel(const float* __restrict__ x, __bf16* __restrict__ xb,
                                unsigned char* __restrict__ mask) {
  int row = blockIdx.x;          // b*T + t
  int tid = threadIdx.x;         // 0..127 (D=128)
  float v = x[(size_t)row * DIN0 + tid];
  xb[(size_t)row * DIN0 + tid] = (__bf16)v;
  __shared__ int anynz;
  if (tid == 0) anynz = 0;
  __syncthreads();
  if (v != 0.f) atomicOr(&anynz, 1);
  __syncthreads();
  if (tid == 0) mask[row] = (unsigned char)anynz;
}

// ---------------------------------------------------------------------------
// Hoisted input projection: out(BT x 1024) = A(BT x KT*32) @ W, bf16 result.
// A and out are streamed once -> non-temporal; weights stay L2-resident.
// ---------------------------------------------------------------------------
template <int KT>
__global__ __launch_bounds__(256) void gemm1024_kernel(
    const __bf16* __restrict__ A,    // (BT, KT*32) row-major
    const __bf16* __restrict__ Wp,   // packed (KT*32 x 1024)
    __bf16* __restrict__ out)        // (BT, 1024)
{
  const int lane = threadIdx.x & 31;
  const int wave = threadIdx.x >> 5;
  const size_t row0 = (size_t)blockIdx.x * 16;
  const int nt0 = wave * 8;

  v8f acc[8];
#pragma unroll
  for (int n = 0; n < 8; ++n) acc[n] = vzero();

  const __bf16* arow = A + row0 * (KT * 32);
#pragma unroll
  for (int kt = 0; kt < KT; ++kt) {
    v16bf a = load_a_nt(arow, KT * 32, kt, lane);
#pragma unroll
    for (int n = 0; n < 8; ++n)
      acc[n] = wmma_bf16(a, load_b(Wp, kt, nt0 + n, 64, lane), acc[n]);
  }

  const int r0 = (lane < 16) ? 0 : 8;
#pragma unroll
  for (int n = 0; n < 8; ++n) {
    int colg = (nt0 + n) * 16 + (lane & 15);
#pragma unroll
    for (int r = 0; r < 8; ++r)
      __builtin_nontemporal_store((__bf16)acc[n][r], out + (row0 + r0 + r) * 1024 + colg);
  }
}

// ---------------------------------------------------------------------------
// Sequential LSTM recurrence (h@Wh only; x@Wx precomputed in Gpre).
// One block = 16 batch rows for all T steps; h,c live in LDS.
// 16 waves x 64 gate columns each. Wh fragments stream from L2 every step.
// ---------------------------------------------------------------------------
__global__ __launch_bounds__(512) void lstm_kernel(
    const __bf16* __restrict__ Gpre,  // (B, T, 1024) precomputed x@Wx, bf16
    const __bf16* __restrict__ Whp,   // packed (256 x 1024)
    const float*  __restrict__ bias,  // (1024) f32
    const unsigned char* __restrict__ mask,  // (B*T)
    __bf16* __restrict__ hout)        // (B, T, 256) bf16
{
  __shared__ float  s_gates[16 * LDG];  // ~66 KB
  __shared__ __bf16 s_h[16 * LDH];      // ~8.3 KB (padded: conflict-free A loads)
  __shared__ float  s_c[16 * 256];      // 16 KB

  const int lane = threadIdx.x & 31;
  const int wave = threadIdx.x >> 5;    // 0..15
  const int b0   = blockIdx.x * 16;
  const int nt0  = wave * 4;            // 4 N-tiles (64 gate columns) per wave

  for (int i = threadIdx.x; i < 16 * LDH; i += 512) s_h[i] = (__bf16)0.f;
  for (int i = threadIdx.x; i < 16 * 256; i += 512) s_c[i] = 0.f;
  __syncthreads();

  for (int t = 0; t < TSTEPS; ++t) {
    // Opaque zero offset, refreshed each step: defeats cross-timestep
    // invariant-load hoisting (which spilled to scratch) while preserving
    // global address-space inference (global_load_b128, LOADcnt only --
    // not flat_load, which would also tie up DScnt / the LDS path).
    int tzero = 0;
    asm volatile("" : "+s"(tzero));
    const __bf16* whp = Whp + tzero;

    v8f acc[4];
#pragma unroll
    for (int n = 0; n < 4; ++n) acc[n] = vzero();

#pragma unroll
    for (int kt = 0; kt < 8; ++kt) {
      v16bf a = load_a(s_h, LDH, kt, lane);
#pragma unroll
      for (int n = 0; n < 4; ++n)
        acc[n] = wmma_bf16(a, load_b(whp, kt, nt0 + n, 64, lane), acc[n]);
    }
#pragma unroll
    for (int n = 0; n < 4; ++n)
      store_c(s_gates + (nt0 + n) * 16, LDG, acc[n], lane);
    __syncthreads();

    // Gate nonlinearities + state update (i,f,g,o chunks of 256)
    for (int i = threadIdx.x; i < 16 * 256; i += 512) {
      int row = i >> 8;
      int col = i & 255;
      const size_t gb = ((size_t)(b0 + row) * TSTEPS + t) * 1024;
      float gi = (float)__builtin_nontemporal_load(Gpre + gb +       col);
      float gf = (float)__builtin_nontemporal_load(Gpre + gb + 256 + col);
      float gg0 = (float)__builtin_nontemporal_load(Gpre + gb + 512 + col);
      float go = (float)__builtin_nontemporal_load(Gpre + gb + 768 + col);
      float zi = s_gates[row * LDG +       col] + gi  + bias[      col];
      float zf = s_gates[row * LDG + 256 + col] + gf  + bias[256 + col];
      float zg = s_gates[row * LDG + 512 + col] + gg0 + bias[512 + col];
      float zo = s_gates[row * LDG + 768 + col] + go  + bias[768 + col];
      float ig = sigm(zi), fg = sigm(zf), og = sigm(zo);
      float gg = tanhf(zg);
      float c_old = s_c[i];
      float h_old = (float)s_h[row * LDH + col];
      float c_new = fg * c_old + ig * gg;
      float h_new = og * tanhf(c_new);
      if (!mask[(size_t)(b0 + row) * TSTEPS + t]) { c_new = c_old; h_new = h_old; }
      s_c[i] = c_new;
      s_h[row * LDH + col] = (__bf16)h_new;
      __builtin_nontemporal_store((__bf16)h_new,
          hout + ((size_t)(b0 + row) * TSTEPS + t) * HID + col);
    }
    __syncthreads();
  }
}

// ---------------------------------------------------------------------------
// Selector MLP: sigmoid(zW1+b1) -> sigmoid(.W2+b2) -> argmax(.Wo+bo)
// ---------------------------------------------------------------------------
__global__ __launch_bounds__(256) void sel_kernel(
    const __bf16* __restrict__ z, const __bf16* __restrict__ W1p, const float* __restrict__ b1,
    const __bf16* __restrict__ W2p, const float* __restrict__ b2,
    const __bf16* __restrict__ Wop, const float* __restrict__ bo,
    int* __restrict__ idxout) {
  __shared__ float  s_f[16 * LDF];
  __shared__ __bf16 s_a[16 * LDH];
  __shared__ float  s_logit[16 * LDL_SEL];
  const int lane = threadIdx.x & 31;
  const int wave = threadIdx.x >> 5;
  const size_t row0 = (size_t)blockIdx.x * 16;

  { // layer 1 (from global z, streamed once)
    v8f acc[2] = {vzero(), vzero()};
    const __bf16* arow = z + row0 * 256;
#pragma unroll
    for (int kt = 0; kt < 8; ++kt) {
      v16bf a = load_a_nt(arow, 256, kt, lane);
#pragma unroll
      for (int n = 0; n < 2; ++n)
        acc[n] = wmma_bf16(a, load_b(W1p, kt, wave * 2 + n, 16, lane), acc[n]);
    }
#pragma unroll
    for (int n = 0; n < 2; ++n) store_c(s_f + (wave * 2 + n) * 16, LDF, acc[n], lane);
  }
  __syncthreads();
  for (int i = threadIdx.x; i < 4096; i += 256) {
    int row = i >> 8, col = i & 255;
    s_a[row * LDH + col] = (__bf16)sigm(s_f[row * LDF + col] + b1[col]);
  }
  __syncthreads();

  { // layer 2 (from LDS)
    v8f acc[2] = {vzero(), vzero()};
#pragma unroll
    for (int kt = 0; kt < 8; ++kt) {
      v16bf a = load_a(s_a, LDH, kt, lane);
#pragma unroll
      for (int n = 0; n < 2; ++n)
        acc[n] = wmma_bf16(a, load_b(W2p, kt, wave * 2 + n, 16, lane), acc[n]);
    }
#pragma unroll
    for (int n = 0; n < 2; ++n) store_c(s_f + (wave * 2 + n) * 16, LDF, acc[n], lane);
  }
  __syncthreads();
  for (int i = threadIdx.x; i < 4096; i += 256) {
    int row = i >> 8, col = i & 255;
    s_a[row * LDH + col] = (__bf16)sigm(s_f[row * LDF + col] + b2[col]);
  }
  __syncthreads();

  // output layer: N = 64 -> waves 0..3
  if (wave < 4) {
    v8f acc = vzero();
#pragma unroll
    for (int kt = 0; kt < 8; ++kt)
      acc = wmma_bf16(load_a(s_a, LDH, kt, lane), load_b(Wop, kt, wave, 4, lane), acc);
    store_c(s_logit + wave * 16, LDL_SEL, acc, lane);
  }
  __syncthreads();

  if (threadIdx.x < 16) {  // argmax(softmax) == argmax(logits); first-max ties
    int r = threadIdx.x;
    float best = s_logit[r * LDL_SEL] + bo[0];
    int bi = 0;
    for (int j = 1; j < NCLUST; ++j) {
      float v = s_logit[r * LDL_SEL + j] + bo[j];
      if (v > best) { best = v; bi = j; }
    }
    idxout[row0 + r] = bi;
  }
}

// ---------------------------------------------------------------------------
// Predictor MLP on gathered embeddings, softmax output (f32)
// ---------------------------------------------------------------------------
__global__ __launch_bounds__(256) void pred_kernel(
    const int* __restrict__ idx, const float* __restrict__ emb,
    const __bf16* __restrict__ W1p, const float* __restrict__ b1,
    const __bf16* __restrict__ W2p, const float* __restrict__ b2,
    const __bf16* __restrict__ Wop, const float* __restrict__ bo,
    float* __restrict__ yout) {
  __shared__ float  s_f[16 * LDF];
  __shared__ __bf16 s_a[16 * LDH];
  __shared__ float  s_logit[16 * LDL_PRE];
  const int lane = threadIdx.x & 31;
  const int wave = threadIdx.x >> 5;
  const size_t row0 = (size_t)blockIdx.x * 16;

  // gather centroid rows -> bf16 LDS (emb is tiny & reused -> regular loads)
  for (int i = threadIdx.x; i < 4096; i += 256) {
    int r = i >> 8, col = i & 255;
    int k = idx[row0 + r];
    s_a[r * LDH + col] = (__bf16)emb[(size_t)k * HID + col];
  }
  __syncthreads();

  { // layer 1
    v8f acc[2] = {vzero(), vzero()};
#pragma unroll
    for (int kt = 0; kt < 8; ++kt) {
      v16bf a = load_a(s_a, LDH, kt, lane);
#pragma unroll
      for (int n = 0; n < 2; ++n)
        acc[n] = wmma_bf16(a, load_b(W1p, kt, wave * 2 + n, 16, lane), acc[n]);
    }
#pragma unroll
    for (int n = 0; n < 2; ++n) store_c(s_f + (wave * 2 + n) * 16, LDF, acc[n], lane);
  }
  __syncthreads();
  for (int i = threadIdx.x; i < 4096; i += 256) {
    int row = i >> 8, col = i & 255;
    s_a[row * LDH + col] = (__bf16)sigm(s_f[row * LDF + col] + b1[col]);
  }
  __syncthreads();

  { // layer 2
    v8f acc[2] = {vzero(), vzero()};
#pragma unroll
    for (int kt = 0; kt < 8; ++kt) {
      v16bf a = load_a(s_a, LDH, kt, lane);
#pragma unroll
      for (int n = 0; n < 2; ++n)
        acc[n] = wmma_bf16(a, load_b(W2p, kt, wave * 2 + n, 16, lane), acc[n]);
    }
#pragma unroll
    for (int n = 0; n < 2; ++n) store_c(s_f + (wave * 2 + n) * 16, LDF, acc[n], lane);
  }
  __syncthreads();
  for (int i = threadIdx.x; i < 4096; i += 256) {
    int row = i >> 8, col = i & 255;
    s_a[row * LDH + col] = (__bf16)sigm(s_f[row * LDF + col] + b2[col]);
  }
  __syncthreads();

  // output layer: N = 32 -> waves 0..1
  if (wave < 2) {
    v8f acc = vzero();
#pragma unroll
    for (int kt = 0; kt < 8; ++kt)
      acc = wmma_bf16(load_a(s_a, LDH, kt, lane), load_b(Wop, kt, wave, 2, lane), acc);
    store_c(s_logit + wave * 16, LDL_PRE, acc, lane);
  }
  __syncthreads();

  if (threadIdx.x < 16) {  // softmax over 32 outputs
    int r = threadIdx.x;
    float l[OUTD];
    float mx = -3.4e38f;
    for (int j = 0; j < OUTD; ++j) { l[j] = s_logit[r * LDL_PRE + j] + bo[j]; mx = fmaxf(mx, l[j]); }
    float s = 0.f;
    for (int j = 0; j < OUTD; ++j) { l[j] = __expf(l[j] - mx); s += l[j]; }
    float inv = 1.f / s;
    for (int j = 0; j < OUTD; ++j) yout[(row0 + r) * OUTD + j] = l[j] * inv;
  }
}

// ---------------------------------------------------------------------------
extern "C" void kernel_launch(void* const* d_in, const int* in_sizes, int n_in,
                              void* d_out, int out_size, void* d_ws, size_t ws_size,
                              hipStream_t stream) {
  (void)in_sizes; (void)n_in; (void)out_size; (void)ws_size;
  const float* x    = (const float*)d_in[0];
  const float* e0Wx = (const float*)d_in[1];
  const float* e0Wh = (const float*)d_in[2];
  const float* e0b  = (const float*)d_in[3];
  const float* e1Wx = (const float*)d_in[4];
  const float* e1Wh = (const float*)d_in[5];
  const float* e1b  = (const float*)d_in[6];
  const float* sW1  = (const float*)d_in[7];
  const float* sb1  = (const float*)d_in[8];
  const float* sW2  = (const float*)d_in[9];
  const float* sb2  = (const float*)d_in[10];
  const float* sWo  = (const float*)d_in[11];
  const float* sbo  = (const float*)d_in[12];
  const float* emb  = (const float*)d_in[13];
  const float* pW1  = (const float*)d_in[14];
  const float* pb1  = (const float*)d_in[15];
  const float* pW2  = (const float*)d_in[16];
  const float* pb2  = (const float*)d_in[17];
  const float* pWo  = (const float*)d_in[18];
  const float* pbo  = (const float*)d_in[19];

  char* w = (char*)d_ws;
  auto carve = [&](size_t bytes) { char* p = w; w += (bytes + 255) & ~(size_t)255; return p; };
  __bf16* xb   = (__bf16*)carve((size_t)BT * DIN0 * 2);        // 16 MB
  __bf16* hb   = (__bf16*)carve((size_t)BT * HID * 2);         // 32 MB
  __bf16* zb   = (__bf16*)carve((size_t)BT * HID * 2);         // 32 MB
  __bf16* gpre = (__bf16*)carve((size_t)BT * 1024 * 2);        // 128 MB (reused L0/L1)
  unsigned char* mask = (unsigned char*)carve(BT);
  __bf16* Wx0p = (__bf16*)carve((size_t)128 * 1024 * 2);
  __bf16* Wh0p = (__bf16*)carve((size_t)256 * 1024 * 2);
  __bf16* Wx1p = (__bf16*)carve((size_t)256 * 1024 * 2);
  __bf16* Wh1p = (__bf16*)carve((size_t)256 * 1024 * 2);
  __bf16* sW1p = (__bf16*)carve((size_t)256 * 256 * 2);
  __bf16* sW2p = (__bf16*)carve((size_t)256 * 256 * 2);
  __bf16* sWop = (__bf16*)carve((size_t)256 * 64 * 2);
  __bf16* pW1p = (__bf16*)carve((size_t)256 * 256 * 2);
  __bf16* pW2p = (__bf16*)carve((size_t)256 * 256 * 2);
  __bf16* pWop = (__bf16*)carve((size_t)256 * 32 * 2);
  int* idxb    = (int*)carve((size_t)BT * 4);

  auto pack = [&](const float* W, __bf16* out, int K, int N) {
    int total = K * N;
    pack_w_kernel<<<(total + 255) / 256, 256, 0, stream>>>(W, out, K, N);
  };

  mask_cvt_kernel<<<BT, 128, 0, stream>>>(x, xb, mask);
  pack(e0Wx, Wx0p, 128, 1024);
  pack(e0Wh, Wh0p, 256, 1024);
  pack(e1Wx, Wx1p, 256, 1024);
  pack(e1Wh, Wh1p, 256, 1024);
  pack(sW1, sW1p, 256, 256);
  pack(sW2, sW2p, 256, 256);
  pack(sWo, sWop, 256, 64);
  pack(pW1, pW1p, 256, 256);
  pack(pW2, pW2p, 256, 256);
  pack(pWo, pWop, 256, 32);

  // Layer 0: hoisted input projection (fully parallel), then serial recurrence
  gemm1024_kernel<4><<<BT / 16, 256, 0, stream>>>(xb, Wx0p, gpre);
  lstm_kernel<<<BATCH / 16, 512, 0, stream>>>(gpre, Wh0p, e0b, mask, hb);
  // Layer 1: same, reusing the gpre buffer
  gemm1024_kernel<8><<<BT / 16, 256, 0, stream>>>(hb, Wx1p, gpre);
  lstm_kernel<<<BATCH / 16, 512, 0, stream>>>(gpre, Wh1p, e1b, mask, zb);

  sel_kernel<<<BT / 16, 256, 0, stream>>>(zb, sW1p, sb1, sW2p, sb2, sWop, sbo, idxb);
  pred_kernel<<<BT / 16, 256, 0, stream>>>(idxb, emb, pW1p, pb1, pW2p, pb2, pWop, pbo,
                                           (float*)d_out);
}
